// SpectralT_Block_68719476736283
// MI455X (gfx1250) — matrix operand order repaired and checked by reading, not compiled, not run on hardware
//
#include <hip/hip_runtime.h>
#include <hip/hip_bf16.h>
#include <math.h>

typedef __attribute__((ext_vector_type(16))) _Float16 v16h;
typedef __attribute__((ext_vector_type(8)))  float    v8f;
typedef __attribute__((ext_vector_type(4)))  int      i32x4;

#define B_    2
#define C_    512
#define H_    128
#define W_    128
#define N_    16384        // H_*W_
#define HEADS 8
#define D_    64
#define CM_   2048
#define BN_   32768        // B_*N_

// gfx1250 async global->LDS path (ASYNCcnt), with safe fallback
#if defined(__has_builtin)
# if __has_builtin(__builtin_amdgcn_global_load_async_to_lds_b128) && \
     __has_builtin(__builtin_amdgcn_s_wait_asynccnt)
#  define USE_ASYNC_LDS 1
# endif
#endif
#ifndef USE_ASYNC_LDS
# define USE_ASYNC_LDS 0
#endif

// ---------------- workspace layout (bytes) ----------------
#define OFF_X    0x00000000ULL   // f32 [B,N,C]      64 MiB
#define OFF_XB   0x04000000ULL   // f16 [B,N,C]      32 MiB
#define OFF_QKV  0x06000000ULL   // f16 [B,N,3C]     96 MiB
#define OFF_O    0x0C000000ULL   // f16 [B,N,C]      32 MiB
#define OFF_G    0x0E000000ULL   // f32 [B,N,C]      64 MiB
#define OFF_PT   0x12000000ULL   // f16 [B,N,C]      32 MiB
#define OFF_XN   0x14000000ULL   // f16 [B,N,C]      32 MiB
#define OFF_Y    0x16000000ULL   // f16 [B,N,CM]    128 MiB
#define OFF_OY   0x1E000000ULL   // f16 [B,N,CM]    128 MiB
#define OFF_WCAT 0x26000000ULL   // f16 [3C,C]  (B^T)
#define OFF_PW   0x26200000ULL   // f16 [C,C]   (B^T)
#define OFF_W1T  0x26300000ULL   // f16 [CM,C]  (B^T)
#define OFF_W2T  0x26500000ULL   // f16 [C,CM]  (B^T)
#define OFF_QINV 0x26700000ULL   // f32 [B*H*64]
#define OFF_KINV 0x26701000ULL   // f32 [B*H*64]
#define OFF_ATTN 0x26702000ULL   // f32 [B*H,64,64]

// =========================================================
//  Tiled WMMA GEMM: C[M,N] = A[M,K] * Bt[N,K]^T   (f16 in, f32 acc)
//  256 threads = 8 waves; block tile 128x128; wave tile 32x64
//  (4x2 wave grid -> 2x4 WMMA tiles per wave); double-buffered LDS,
//  async global->LDS when available.
// =========================================================
__global__ __launch_bounds__(256)
void k_gemm_f16(const _Float16* __restrict__ A,   // [M,K] row-major
                const _Float16* __restrict__ Bt,  // [N,K] row-major (transposed B)
                float* __restrict__ Cf32,         // optional f32 out
                _Float16* __restrict__ Cf16,      // else f16 out
                const float* __restrict__ bias,   // optional [N]
                int M, int N, int K)
{
    __shared__ _Float16 sA[2][128][40];   // [buf][m][k] padded row (80B)
    __shared__ _Float16 sB[2][128][40];   // [buf][n][k] padded row

    const int tid  = threadIdx.x;
    const int wave = tid >> 5;
    const int lane = tid & 31;
    const int wm0  = (wave >> 1) * 32;    // wave row offset (4 rows of waves)
    const int wn0  = (wave & 1) * 64;     // wave col offset (2 cols of waves)
    const int m0   = blockIdx.y * 128;
    const int n0   = blockIdx.x * 128;

    // staging: each thread owns a 16-half (32B) run: row = tid/2, cols scol..scol+15
    const int srow = tid >> 1;
    const int scol = (tid & 1) * 16;
    const _Float16* aBase = A  + (size_t)(m0 + srow) * K + scol;
    const _Float16* bBase = Bt + (size_t)(n0 + srow) * K + scol;

    const int KT = K >> 5;
    v8f acc[2][4] = {};

    auto stage = [&](int kt, int buf) {
        const int k0 = kt << 5;
#if USE_ASYNC_LDS
        __builtin_amdgcn_global_load_async_to_lds_b128(
            (i32x4*)(aBase + k0),     (i32x4*)&sA[buf][srow][scol],     0, 0);
        __builtin_amdgcn_global_load_async_to_lds_b128(
            (i32x4*)(aBase + k0 + 8), (i32x4*)&sA[buf][srow][scol + 8], 0, 0);
        __builtin_amdgcn_global_load_async_to_lds_b128(
            (i32x4*)(bBase + k0),     (i32x4*)&sB[buf][srow][scol],     0, 0);
        __builtin_amdgcn_global_load_async_to_lds_b128(
            (i32x4*)(bBase + k0 + 8), (i32x4*)&sB[buf][srow][scol + 8], 0, 0);
#else
        const float4 a0 = *(const float4*)(aBase + k0);
        const float4 a1 = *(const float4*)(aBase + k0 + 8);
        const float4 b0 = *(const float4*)(bBase + k0);
        const float4 b1 = *(const float4*)(bBase + k0 + 8);
        *(float4*)&sA[buf][srow][scol]     = a0;
        *(float4*)&sA[buf][srow][scol + 8] = a1;
        *(float4*)&sB[buf][srow][scol]     = b0;
        *(float4*)&sB[buf][srow][scol + 8] = b1;
#endif
    };

    stage(0, 0);
#if USE_ASYNC_LDS
    __builtin_amdgcn_s_wait_asynccnt(0);
#endif
    __syncthreads();

    const int arow  = lane & 15;
    const int khalf = lane >> 4;
    const int kb    = khalf * 16;

    for (int kt = 0; kt < KT; kt++) {
        const int cur = kt & 1;
        if (kt + 1 < KT) stage(kt + 1, cur ^ 1);
        if (kt + 2 < KT) {   // gfx1250 global_prefetch of the tile after next
            __builtin_prefetch(aBase + ((kt + 2) << 5), 0, 1);
            __builtin_prefetch(bBase + ((kt + 2) << 5), 0, 1);
        }

        // fragment loads: all contiguous 16B LDS reads (ds_load_b128)
        union { v16h v; float4 q[2]; } af[2], bfr[4];
        #pragma unroll
        for (int i = 0; i < 2; i++) {
            const _Float16* ap = &sA[cur][wm0 + i * 16 + arow][0];
            af[i].q[0] = *(const float4*)(ap + khalf * 8);        // K 0..7 | 8..15
            af[i].q[1] = *(const float4*)(ap + 16 + khalf * 8);   // K 16..23 | 24..31
        }
        #pragma unroll
        for (int j = 0; j < 4; j++) {
            const _Float16* bp = &sB[cur][wn0 + j * 16 + arow][0];
            bfr[j].q[0] = *(const float4*)(bp + khalf * 8);       // K 0..7 | 8..15
            bfr[j].q[1] = *(const float4*)(bp + 16 + khalf * 8);  // K 16..23 | 24..31
        }
        #pragma unroll
        for (int i = 0; i < 2; i++)
            #pragma unroll
            for (int j = 0; j < 4; j++)
                acc[i][j] = __builtin_amdgcn_wmma_f32_16x16x32_f16(
                    false, af[i].v, false, bfr[j].v, (short)0, acc[i][j], false, false);

#if USE_ASYNC_LDS
        if (kt + 1 < KT) __builtin_amdgcn_s_wait_asynccnt(0);
#endif
        __syncthreads();
    }

    // epilogue: C/D layout lane<16 -> M=r, lane>=16 -> M=8+r
    const int ccol  = lane & 15;
    const int rbase = (lane >> 4) * 8;
    for (int i = 0; i < 2; i++)
        for (int j = 0; j < 4; j++) {
            const int gm = m0 + wm0 + i * 16 + rbase;
            const int gn = n0 + wn0 + j * 16 + ccol;
            #pragma unroll
            for (int r = 0; r < 8; r++) {
                float v = acc[i][j][r];
                if (bias) v += bias[gn];
                if (Cf32) Cf32[(size_t)(gm + r) * N + gn] = v;
                else      Cf16[(size_t)(gm + r) * N + gn] = (_Float16)v;
            }
        }
}

// =========================================================
//  attn[b,h,d,e] = sum_n k~[d,n] * q~[e,n]   (WMMA over n)
//  d-major LDS staging so fragments are contiguous b128 reads
// =========================================================
__global__ __launch_bounds__(256)
void k_attn_qk(const _Float16* __restrict__ QKV,
               const float* __restrict__ qinv, const float* __restrict__ kinv,
               const float* __restrict__ rescale,   // [HEADS] for this layer
               float* __restrict__ attn)            // [B*H,64,64]
{
    __shared__ _Float16 sKT[64][40];   // [d][n-local]
    __shared__ _Float16 sQT[64][40];   // [e][n-local]

    const int tid  = threadIdx.x;
    const int wave = tid >> 5;
    const int lane = tid & 31;
    const int bh   = blockIdx.x;
    const int b    = bh >> 3, h = bh & 7;
    const int tm   = wave & 3;
    const int tn0  = (wave >> 2) * 2;

    const int nl = (tid * 8) >> 6;   // staging n-local row (0..31)
    const int d0 = (tid * 8) & 63;   // staging d base
    const float* qi = qinv + bh * 64;
    const float* ki = kinv + bh * 64;

    const int arow  = lane & 15;
    const int khalf = lane >> 4;
    const int kb    = khalf * 16;
    const int drow  = tm * 16 + arow;

    v8f acc[2] = {};

    for (int n0 = 0; n0 < N_; n0 += 32) {
        const size_t base = ((size_t)b * N_ + n0 + nl) * (3 * C_) + h * D_;
        const _Float16* qp = QKV + base;        // q at channel offset 0
        const _Float16* kp = QKV + base + C_;   // k at channel offset 512
        #pragma unroll
        for (int e = 0; e < 8; e++) {
            sQT[d0 + e][nl] = (_Float16)((float)qp[d0 + e] * qi[d0 + e]);
            sKT[d0 + e][nl] = (_Float16)((float)kp[d0 + e] * ki[d0 + e]);
        }
        __syncthreads();

        union { v16h v; float4 q[2]; } a, bb[2];
        a.q[0] = *(const float4*)(&sKT[drow][0] + khalf * 8);
        a.q[1] = *(const float4*)(&sKT[drow][0] + 16 + khalf * 8);
        #pragma unroll
        for (int j = 0; j < 2; j++) {
            const _Float16* bp = &sQT[(tn0 + j) * 16 + arow][0];
            bb[j].q[0] = *(const float4*)(bp + khalf * 8);
            bb[j].q[1] = *(const float4*)(bp + 16 + khalf * 8);
            acc[j] = __builtin_amdgcn_wmma_f32_16x16x32_f16(
                false, a.v, false, bb[j].v, (short)0, acc[j], false, false);
        }
        __syncthreads();
    }

    const float rsc = rescale[h];
    const int ccol  = lane & 15;
    const int rbase = (lane >> 4) * 8;
    for (int j = 0; j < 2; j++) {
        const int eg = (tn0 + j) * 16 + ccol;
        #pragma unroll
        for (int r = 0; r < 8; r++) {
            const int dg = tm * 16 + rbase + r;
            attn[(size_t)bh * 4096 + dg * 64 + eg] = acc[j][r] * rsc;
        }
    }
}

// ----- l2-norm inverse scales over n for q and k rows -----
__global__ __launch_bounds__(256)
void k_qknorms(const _Float16* __restrict__ QKV,
               float* __restrict__ qinv, float* __restrict__ kinv)
{
    const int bhd = blockIdx.x;          // b*512 + h*64 + d
    const int b   = bhd >> 9;
    const int hd  = bhd & 511;
    const _Float16* qp = QKV + (size_t)b * N_ * (3 * C_) + hd;
    const _Float16* kp = qp + C_;
    float sq = 0.f, sk = 0.f;
    for (int n = threadIdx.x; n < N_; n += 256) {
        float q = (float)qp[(size_t)n * (3 * C_)]; sq += q * q;
        float k = (float)kp[(size_t)n * (3 * C_)]; sk += k * k;
    }
    __shared__ float rq[256], rk[256];
    rq[threadIdx.x] = sq; rk[threadIdx.x] = sk;
    __syncthreads();
    for (int s = 128; s > 0; s >>= 1) {
        if (threadIdx.x < s) { rq[threadIdx.x] += rq[threadIdx.x + s];
                               rk[threadIdx.x] += rk[threadIdx.x + s]; }
        __syncthreads();
    }
    if (threadIdx.x == 0) {
        qinv[bhd] = 1.0f / fmaxf(sqrtf(rq[0]), 1e-12f);
        kinv[bhd] = 1.0f / fmaxf(sqrtf(rk[0]), 1e-12f);
    }
}

// ----- softmax over last axis (64 wide); one wave32 per row -----
__global__ __launch_bounds__(256)
void k_softmax64(float* __restrict__ attn)
{
    const int row  = blockIdx.x * 8 + (threadIdx.x >> 5);
    const int lane = threadIdx.x & 31;
    float* rp = attn + (size_t)row * 64;
    float v0 = rp[lane], v1 = rp[lane + 32];
    float m = fmaxf(v0, v1);
    for (int s = 16; s > 0; s >>= 1) m = fmaxf(m, __shfl_xor(m, s, 32));
    float e0 = __expf(v0 - m), e1 = __expf(v1 - m);
    float s = e0 + e1;
    for (int ss = 16; ss > 0; ss >>= 1) s += __shfl_xor(s, ss, 32);
    rp[lane] = e0 / s; rp[lane + 32] = e1 / s;
}

// ----- o[b,n,h*64+d] = sum_e attn[b,h,d,e] * v[b,n,h*64+e] -----
__global__ __launch_bounds__(256)
void k_attn_apply(const float* __restrict__ attn,
                  const _Float16* __restrict__ QKV,
                  _Float16* __restrict__ O)
{
    const int blk   = blockIdx.x;
    const int bh    = blk >> 6;       // 64 chunks per (b,h)
    const int chunk = blk & 63;
    __shared__ float sAtt[64][64 + 1];
    for (int i = threadIdx.x; i < 4096; i += 256)
        sAtt[i >> 6][i & 63] = attn[(size_t)bh * 4096 + i];
    __syncthreads();
    const int b = bh >> 3, h = bh & 7;
    const int d = threadIdx.x & 63;
    const int pl = threadIdx.x >> 6;  // 0..3
    for (int n = chunk * 256 + pl; n < chunk * 256 + 256; n += 4) {
        const _Float16* vp = QKV + ((size_t)b * N_ + n) * (3 * C_) + 2 * C_ + h * D_;
        float acc = 0.f;
        #pragma unroll 8
        for (int e = 0; e < 64; e++) acc += sAtt[d][e] * (float)vp[e];
        O[((size_t)b * N_ + n) * C_ + h * D_ + d] = (_Float16)acc;
    }
}

// ----- channel-last depthwise conv over a 512-channel slice -----
__global__ __launch_bounds__(256)
void k_dwconv(const _Float16* __restrict__ in, int inC, int inOff,
              const float* __restrict__ w, int ks, int pad,
              float* __restrict__ outF32, _Float16* __restrict__ outF16,
              const _Float16* __restrict__ resid,
              int outC, int outOff, int doGelu)
{
    const int idx = blockIdx.x * 256 + threadIdx.x;   // over B*N*512
    const int ch  = idx & 511;
    const int pix = (idx >> 9) & 16383;
    const int b   = idx >> 23;
    const int py = pix >> 7, px = pix & 127;
    float acc = 0.f;
    const float* wc = w + ch * ks * ks;
    for (int dy = 0; dy < ks; dy++) {
        const int y = py + dy - pad;
        if ((unsigned)y >= (unsigned)H_) continue;
        for (int dx = 0; dx < ks; dx++) {
            const int x = px + dx - pad;
            if ((unsigned)x >= (unsigned)W_) continue;
            acc += wc[dy * ks + dx] *
                   (float)in[(((size_t)b * N_) + (y << 7) + x) * inC + inOff + ch];
        }
    }
    if (doGelu) acc = 0.5f * acc * (1.0f + erff(acc * 0.70710678118f));
    const size_t oidx = (((size_t)b * N_) + pix) * outC + outOff + ch;
    if (resid) acc += (float)resid[oidx];
    if (outF32) outF32[oidx] += acc;
    else        outF16[oidx] = (_Float16)acc;
}

// ----- LayerNorm over channel axis, f16 out -----
__global__ __launch_bounds__(256)
void k_layernorm(const float* __restrict__ X,
                 const float* __restrict__ g, const float* __restrict__ bta,
                 _Float16* __restrict__ XN)
{
    const int p = blockIdx.x;                 // b*N + pix
    const float* xp = X + (size_t)p * C_;
    float s = 0.f, s2 = 0.f;
    for (int c = threadIdx.x; c < C_; c += 256) {
        float v = xp[c]; s += v; s2 += v * v;
    }
    __shared__ float bs[256], bs2[256];
    bs[threadIdx.x] = s; bs2[threadIdx.x] = s2;
    __syncthreads();
    for (int st = 128; st > 0; st >>= 1) {
        if (threadIdx.x < st) { bs[threadIdx.x] += bs[threadIdx.x + st];
                                bs2[threadIdx.x] += bs2[threadIdx.x + st]; }
        __syncthreads();
    }
    const float mu   = bs[0] * (1.0f / C_);
    const float var  = bs2[0] * (1.0f / C_) - mu * mu;
    const float rstd = rsqrtf(var + 1e-5f);
    for (int c = threadIdx.x; c < C_; c += 256)
        XN[(size_t)p * C_ + c] = (_Float16)((xp[c] - mu) * rstd * g[c] + bta[c]);
}

// ----- residual update: X += G; refresh f16 shadow -----
__global__ __launch_bounds__(256)
void k_resadd(float* __restrict__ X, const float* __restrict__ G,
              _Float16* __restrict__ XB)
{
    const size_t i = (size_t)blockIdx.x * 256 + threadIdx.x;
    const float v = X[i] + G[i];
    X[i] = v;
    XB[i] = (_Float16)v;
}

// ----- layout converters -----
__global__ __launch_bounds__(256)
void k_nchw_to_nlc(const float* __restrict__ src, float* __restrict__ X,
                   _Float16* __restrict__ XB)
{
    const int i   = blockIdx.x * 256 + threadIdx.x;   // dst order [b,pix,ch]
    const int ch  = i & 511;
    const int pix = (i >> 9) & 16383;
    const int b   = i >> 23;
    const float v = src[((size_t)b * C_ + ch) * N_ + pix];
    X[i] = v; XB[i] = (_Float16)v;
}

__global__ __launch_bounds__(256)
void k_nlc_to_nchw(const float* __restrict__ X, float* __restrict__ dst)
{
    const int i   = blockIdx.x * 256 + threadIdx.x;   // dst order [b,ch,pix]
    const int pix = i & 16383;
    const int ch  = (i >> 14) & 511;
    const int b   = i >> 23;
    dst[i] = X[(((size_t)b * N_) + pix) * C_ + ch];
}

__global__ __launch_bounds__(256)
void k_cvt(const float* __restrict__ src, _Float16* __restrict__ dst,
           int rows, int cols, int ldd, int coloff)
{
    const int i = blockIdx.x * 256 + threadIdx.x;
    if (i < rows * cols) {
        const int r = i / cols, c = i % cols;
        dst[(size_t)r * ldd + coloff + c] = (_Float16)src[i];
    }
}

// src [rows,cols] -> dst[(rowoff+c)*ldd + r]   (transpose with row offset)
__global__ __launch_bounds__(256)
void k_cvtT(const float* __restrict__ src, _Float16* __restrict__ dst,
            int rows, int cols, int ldd, int rowoff)
{
    const int i = blockIdx.x * 256 + threadIdx.x;
    if (i < rows * cols) {
        const int r = i / cols, c = i % cols;
        dst[(size_t)(rowoff + c) * ldd + r] = (_Float16)src[i];
    }
}

// =========================================================
extern "C" void kernel_launch(void* const* d_in, const int* in_sizes, int n_in,
                              void* d_out, int out_size, void* d_ws, size_t ws_size,
                              hipStream_t stream)
{
    (void)in_sizes; (void)n_in; (void)out_size; (void)ws_size;
    const float* xin     = (const float*)d_in[0];
    const float* Wq      = (const float*)d_in[1];
    const float* Wk      = (const float*)d_in[2];
    const float* Wv      = (const float*)d_in[3];
    const float* rescale = (const float*)d_in[4];
    const float* projW   = (const float*)d_in[5];
    const float* projB   = (const float*)d_in[6];
    const float* posw1   = (const float*)d_in[7];
    const float* posw2   = (const float*)d_in[8];
    const float* lng     = (const float*)d_in[9];
    const float* lnb     = (const float*)d_in[10];
    const float* w1      = (const float*)d_in[11];
    const float* dw1     = (const float*)d_in[12];
    const float* dw3     = (const float*)d_in[13];
    const float* dw5     = (const float*)d_in[14];
    const float* dw7     = (const float*)d_in[15];
    const float* w2      = (const float*)d_in[16];

    char* ws = (char*)d_ws;
    float*    X    = (float*)   (ws + OFF_X);
    _Float16* XB   = (_Float16*)(ws + OFF_XB);
    _Float16* QKV  = (_Float16*)(ws + OFF_QKV);
    _Float16* O    = (_Float16*)(ws + OFF_O);
    float*    G    = (float*)   (ws + OFF_G);
    _Float16* PT   = (_Float16*)(ws + OFF_PT);
    _Float16* XN   = (_Float16*)(ws + OFF_XN);
    _Float16* Y    = (_Float16*)(ws + OFF_Y);
    _Float16* OY   = (_Float16*)(ws + OFF_OY);
    _Float16* WCAT = (_Float16*)(ws + OFF_WCAT);
    _Float16* PW   = (_Float16*)(ws + OFF_PW);
    _Float16* W1T  = (_Float16*)(ws + OFF_W1T);
    _Float16* W2T  = (_Float16*)(ws + OFF_W2T);
    float*    QINV = (float*)   (ws + OFF_QINV);
    float*    KINV = (float*)   (ws + OFF_KINV);
    float*    ATTN = (float*)   (ws + OFF_ATTN);

    const int ELB = (B_ * N_ * C_) / 256;   // 65536 blocks over b,n,512

    k_nchw_to_nlc<<<ELB, 256, 0, stream>>>(xin, X, XB);

    for (int l = 0; l < 2; l++) {
        // ---- weight conversion: all GEMM B matrices stored as B^T [N,K] f16 ----
        k_cvtT<<<1024, 256, 0, stream>>>(Wq + (size_t)l * C_ * C_, WCAT, C_, C_, C_, 0);
        k_cvtT<<<1024, 256, 0, stream>>>(Wk + (size_t)l * C_ * C_, WCAT, C_, C_, C_, C_);
        k_cvtT<<<1024, 256, 0, stream>>>(Wv + (size_t)l * C_ * C_, WCAT, C_, C_, C_, 2 * C_);
        k_cvtT<<<1024, 256, 0, stream>>>(projW + (size_t)l * C_ * C_, PW, C_, C_, C_, 0);
        k_cvt <<<4096, 256, 0, stream>>>(w1 + (size_t)l * CM_ * C_, W1T, CM_, C_, C_, 0);
        k_cvt <<<4096, 256, 0, stream>>>(w2 + (size_t)l * C_ * CM_, W2T, C_, CM_, CM_, 0);

        // ---- S_MSA ----
        k_gemm_f16<<<dim3(12, 256), 256, 0, stream>>>(XB, WCAT, nullptr, QKV, nullptr,
                                                      BN_, 3 * C_, C_);
        k_qknorms<<<1024, 256, 0, stream>>>(QKV, QINV, KINV);
        k_attn_qk<<<B_ * HEADS, 256, 0, stream>>>(QKV, QINV, KINV,
                                                  rescale + l * HEADS, ATTN);
        k_softmax64<<<128, 256, 0, stream>>>(ATTN);
        k_attn_apply<<<B_ * HEADS * 64, 256, 0, stream>>>(ATTN, QKV, O);
        k_gemm_f16<<<dim3(4, 256), 256, 0, stream>>>(O, PW, G, nullptr,
                                                     projB + l * C_, BN_, C_, C_);
        // positional branch on v slice of QKV
        k_dwconv<<<ELB, 256, 0, stream>>>(QKV, 3 * C_, 2 * C_, posw1 + (size_t)l * C_ * 9,
                                          3, 1, nullptr, PT, nullptr, C_, 0, 1);
        k_dwconv<<<ELB, 256, 0, stream>>>(PT, C_, 0, posw2 + (size_t)l * C_ * 9,
                                          3, 1, G, nullptr, nullptr, C_, 0, 0);
        k_resadd<<<ELB, 256, 0, stream>>>(X, G, XB);

        // ---- PreNorm + MS_FFN ----
        k_layernorm<<<BN_, 256, 0, stream>>>(X, lng + l * C_, lnb + l * C_, XN);
        k_gemm_f16<<<dim3(16, 256), 256, 0, stream>>>(XN, W1T, nullptr, Y, nullptr,
                                                      BN_, CM_, C_);
        k_dwconv<<<ELB, 256, 0, stream>>>(Y, CM_, 0 * C_, dw1 + (size_t)l * C_ * 1,
                                          1, 0, nullptr, OY, Y, CM_, 0 * C_, 0);
        k_dwconv<<<ELB, 256, 0, stream>>>(Y, CM_, 1 * C_, dw3 + (size_t)l * C_ * 9,
                                          3, 1, nullptr, OY, Y, CM_, 1 * C_, 0);
        k_dwconv<<<ELB, 256, 0, stream>>>(Y, CM_, 2 * C_, dw5 + (size_t)l * C_ * 25,
                                          5, 2, nullptr, OY, Y, CM_, 2 * C_, 0);
        k_dwconv<<<ELB, 256, 0, stream>>>(Y, CM_, 3 * C_, dw7 + (size_t)l * C_ * 49,
                                          7, 3, nullptr, OY, Y, CM_, 3 * C_, 0);
        k_gemm_f16<<<dim3(4, 256), 256, 0, stream>>>(OY, W2T, G, nullptr, nullptr,
                                                     BN_, C_, CM_);
        k_resadd<<<ELB, 256, 0, stream>>>(X, G, XB);
    }

    k_nlc_to_nchw<<<ELB, 256, 0, stream>>>(X, (float*)d_out);
}
